// TDsemi2UKsim_46866683134133
// MI455X (gfx1250) — compile-verified, matrix operand order checked by reading, not run on hardware
//
#include <hip/hip_runtime.h>
#include <hip/hip_bf16.h>

#define B_DIM 1024
#define CIN   6
#define THW   2430
#define KCH   5
#define CB    10
#define FEA   128
#define NROW  100000
#define CHUNK 256
#define LDP   (CHUNK + 4)   // LDS row stride (floats): 260 % 64 == 4 -> conflict-free row fetch

// workspace layout (float offsets)
#define WS_W     0      // 30: normalized conv weights [5][6]
#define WS_SSY   32     // 5 : global sum-of-squares per y column
#define WS_SSP   48     // 10: global sum-of-squares per pair column
#define WS_LOSS  64     // 1 : total loss accumulator
#define WS_CN    128    // 1024: normalized centers [8][128]
#define WS_COVD  2048   // 1024*10: raw dg Gram off-diagonals
#define WS_COVB  16384  // 1024*45: raw bbs Gram off-diagonals  (~250 KB total ws)

typedef __attribute__((ext_vector_type(2)))  float    v2f;
typedef __attribute__((ext_vector_type(8)))  float    v8f;
typedef __attribute__((ext_vector_type(16))) _Float16 v16h;

__device__ constexpr int PI5[10] = {0,0,0,0,1,1,1,2,2,3};
__device__ constexpr int PJ5[10] = {1,2,3,4,2,3,4,3,4,4};

#if __has_builtin(__builtin_amdgcn_wmma_f32_16x16x4_f32)
#define HAVE_WMMA_F32 1
#else
#define HAVE_WMMA_F32 0
#endif

// Accumulate Gram of an nk x 32 LDS tile slice (columns [wavebase, wavebase+32))
// into a per-wave 16x16 f32 WMMA accumulator.  A fragment: lane L holds row
// m = L&15; f32 16x16x4 layout: v0 = K(hi*2), v1 = K(hi*2+1), hi = L>>4.  For a
// symmetric product the B fragment equals the A fragment (transpose-symmetric
// B layout), so D += A x A^T.
__device__ inline void gram_accum(const float* lds, int stride, int nk, int wavebase, v8f& acc) {
  const int lane  = threadIdx.x & 31;
  const int m     = lane & 15;
  const bool valid = m < nk;
  const float* rowp = lds + m * stride;
#if HAVE_WMMA_F32
  const int khalf = (lane >> 4) * 2;
  #pragma unroll
  for (int it = 0; it < 8; ++it) {
    const int n0 = wavebase + it * 4 + khalf;
    v2f a;
    a.x = valid ? rowp[n0]     : 0.0f;
    a.y = valid ? rowp[n0 + 1] : 0.0f;
    acc = __builtin_amdgcn_wmma_f32_16x16x4_f32(false, a, false, a, (short)0, acc, false, false);
  }
#else
  const int hi8 = (lane >> 4) * 8;
  v16h a;
  #pragma unroll
  for (int h = 0; h < 16; ++h) {
    const int j = h >> 1, slot = h & 1;
    const int k = ((j & 4) ? 16 : 0) + 2 * (j & 3) + slot + hi8;
    a[h] = valid ? (_Float16)rowp[wavebase + k] : (_Float16)0.0f;
  }
  acc = __builtin_amdgcn_wmma_f32_16x16x32_f16(false, a, false, a, (short)0, acc, false, false);
#endif
}

__device__ inline void pair10(int p, int& c, int& d) {
  int idx = p, cc = 0;
  while (idx >= 9 - cc) { idx -= 9 - cc; ++cc; }
  c = cc; d = cc + 1 + idx;
}

// ---------------- init: zero accumulators, build w and normalized centers --------
__global__ __launch_bounds__(256) void k_init(const float* __restrict__ conv_w,
                                              const float* __restrict__ gate,
                                              const float* __restrict__ centerf,
                                              float* __restrict__ ws) {
  const int tid = threadIdx.x;
  if (tid < 64) ws[32 + tid] = 0.0f;  // covers WS_SSY, WS_SSP, WS_LOSS
  if (tid < KCH) {
    float wr[CIN]; float ss = 0.0f;
    #pragma unroll
    for (int c = 0; c < CIN; ++c) {
      const float gt = gate[tid * CIN + c];
      const float g  = gt > 0.0f ? 1.0f : (gt < 0.0f ? 0.0f : 0.5f); // (sign+1)/2
      const float D  = (c == tid) ? 1.0f : 0.0f;
      const float U  = (1.0f - g) * D + g;
      const float v  = conv_w[tid * CIN + c] * U;
      wr[c] = v; ss += v * v;
    }
    const float inv = 1.0f / fmaxf(sqrtf(ss), 1e-12f);
    #pragma unroll
    for (int c = 0; c < CIN; ++c) ws[WS_W + tid * CIN + c] = wr[c] * inv;
  }
  // centers: one wave per class row (8 waves exactly)
  const int wv = tid >> 5, lane = tid & 31;
  const float* cr = centerf + wv * FEA;
  float v[4]; float ss = 0.0f;
  #pragma unroll
  for (int q = 0; q < 4; ++q) { v[q] = cr[lane + 32 * q]; ss += v[q] * v[q]; }
  #pragma unroll
  for (int mk = 16; mk >= 1; mk >>= 1) ss += __shfl_xor(ss, mk, 32);
  const float inv = 1.0f / fmaxf(sqrtf(ss), 1e-12f);
  #pragma unroll
  for (int q = 0; q < 4; ++q) ws[WS_CN + wv * FEA + lane + 32 * q] = v[q] * inv;
}

// ---------------- pass1: y = x @ w^T, global col sumsq, raw dg Gram (WMMA) -------
__global__ __launch_bounds__(256) void k_pass1(const float* __restrict__ x,
                                               float* __restrict__ ws) {
  __shared__ float s_w[32];
  __shared__ float s_y[KCH][LDP];
  __shared__ float s_red[8 * 256];
  const int tid = threadIdx.x, b = blockIdx.x;
  const int lane = tid & 31, wv = tid >> 5;
  if (tid < KCH * CIN) s_w[tid] = ws[WS_W + tid];
  __syncthreads();
  const float* xb = x + (size_t)b * CIN * THW;
  float acc[KCH];
  #pragma unroll
  for (int k = 0; k < KCH; ++k) acc[k] = 0.0f;
  v8f cg;
  #pragma unroll
  for (int r = 0; r < 8; ++r) cg[r] = 0.0f;

  for (int chunk = 0; chunk < (THW + CHUNK - 1) / CHUNK; ++chunk) {
    const int n  = chunk * CHUNK + tid;
    const int nc = n < THW ? n : THW - 1;          // clamp: loads are unconditional
    const float msk = n < THW ? 1.0f : 0.0f;       // mask applied in VALU, no EXEC dance
    float xi[CIN];
    #pragma unroll
    for (int c = 0; c < CIN; ++c) xi[c] = xb[c * THW + nc];
    #pragma unroll
    for (int k = 0; k < KCH; ++k) {
      float yv = 0.0f;
      #pragma unroll
      for (int c = 0; c < CIN; ++c) yv += xi[c] * s_w[k * CIN + c];
      yv *= msk;
      s_y[k][tid] = yv;
      acc[k] += yv * yv;
    }
    __syncthreads();
    gram_accum(&s_y[0][0], LDP, KCH, wv * 32, cg);
    __syncthreads();
  }
  // reduce 8 per-wave Gram accumulators; C layout: (M = r + 8*(L>>4), N = L&15)
  #pragma unroll
  for (int r = 0; r < 8; ++r) s_red[wv * 256 + r * 32 + lane] = cg[r];
  __syncthreads();
  if (tid < 10) {
    const int c = PI5[tid], d = PJ5[tid];
    float s = 0.0f;
    for (int w = 0; w < 8; ++w) s += s_red[w * 256 + (c & 7) * 32 + (c >> 3) * 16 + d];
    ws[WS_COVD + b * 10 + tid] = s;
  }
  __syncthreads();
  for (int k = 0; k < KCH; ++k) {
    s_red[tid] = acc[k];
    __syncthreads();
    for (int off = 128; off > 0; off >>= 1) {
      if (tid < off) s_red[tid] += s_red[tid + off];
      __syncthreads();
    }
    if (tid == 0) atomicAdd(&ws[WS_SSY + k], s_red[0]);
    __syncthreads();
  }
}

// ---------------- dg whitening loss ---------------------------------------------
__global__ __launch_bounds__(256) void k_dgloss(float* __restrict__ ws) {
  const int b = blockIdx.x * blockDim.x + threadIdx.x;
  if (b >= B_DIM) return;
  float inv[KCH];
  #pragma unroll
  for (int k = 0; k < KCH; ++k) inv[k] = 1.0f / fmaxf(sqrtf(ws[WS_SSY + k]), 1e-6f);
  float s = 0.0f;
  #pragma unroll
  for (int p = 0; p < 10; ++p)
    s += fabsf(ws[WS_COVD + b * 10 + p] * inv[PI5[p]] * inv[PJ5[p]]);
  s = s / (float)(THW - 1) / 10.0f;
  atomicAdd(&ws[WS_LOSS], s / (float)B_DIM);
}

// ---------------- pass2: pairs -> raw out, pair sumsq, raw bbs Gram (WMMA) -------
__global__ __launch_bounds__(256) void k_pass2(const float* __restrict__ x,
                                               float* __restrict__ ws,
                                               float* __restrict__ out) {
  __shared__ float s_w[32];
  __shared__ float s_inv[8];
  __shared__ float s_p[CB][LDP];
  __shared__ float s_red[8 * 256];
  const int tid = threadIdx.x, b = blockIdx.x;
  const int lane = tid & 31, wv = tid >> 5;
  if (tid < KCH * CIN) s_w[tid] = ws[WS_W + tid];
  if (tid < KCH) s_inv[tid] = 1.0f / fmaxf(sqrtf(ws[WS_SSY + tid]), 1e-6f);
  __syncthreads();
  const float* xb = x + (size_t)b * CIN * THW;
  float* ob = out + (size_t)b * CB * THW;
  float acc[CB];
  #pragma unroll
  for (int p = 0; p < CB; ++p) acc[p] = 0.0f;
  v8f cg;
  #pragma unroll
  for (int r = 0; r < 8; ++r) cg[r] = 0.0f;

  for (int chunk = 0; chunk < (THW + CHUNK - 1) / CHUNK; ++chunk) {
    const int n  = chunk * CHUNK + tid;
    const int nc = n < THW ? n : THW - 1;
    const float msk = n < THW ? 1.0f : 0.0f;
    float xi[CIN];
    #pragma unroll
    for (int c = 0; c < CIN; ++c) xi[c] = xb[c * THW + nc];
    float xd[KCH];
    #pragma unroll
    for (int k = 0; k < KCH; ++k) {
      float yv = 0.0f;
      #pragma unroll
      for (int c = 0; c < CIN; ++c) yv += xi[c] * s_w[k * CIN + c];
      xd[k] = yv * s_inv[k];
    }
    float pv[CB];
    #pragma unroll
    for (int p = 0; p < CB; ++p) {
      pv[p] = (xd[PI5[p]] / (xd[PJ5[p]] + 1e-6f)) * msk;
      s_p[p][tid] = pv[p];
      acc[p] += pv[p] * pv[p];
    }
    if (n < THW) {                       // single EXEC guard around the 10 stores
      #pragma unroll
      for (int p = 0; p < CB; ++p) ob[p * THW + n] = pv[p];  // raw; scaled in k_scale
    }
    __syncthreads();
    gram_accum(&s_p[0][0], LDP, CB, wv * 32, cg);
    __syncthreads();
  }
  #pragma unroll
  for (int r = 0; r < 8; ++r) s_red[wv * 256 + r * 32 + lane] = cg[r];
  __syncthreads();
  if (tid < 45) {
    int c, d; pair10(tid, c, d);
    float s = 0.0f;
    for (int w = 0; w < 8; ++w) s += s_red[w * 256 + (c & 7) * 32 + (c >> 3) * 16 + d];
    ws[WS_COVB + b * 45 + tid] = s;
  }
  __syncthreads();
  for (int p = 0; p < CB; ++p) {
    s_red[tid] = acc[p];
    __syncthreads();
    for (int off = 128; off > 0; off >>= 1) {
      if (tid < off) s_red[tid] += s_red[tid + off];
      __syncthreads();
    }
    if (tid == 0) atomicAdd(&ws[WS_SSP + p], s_red[0]);
    __syncthreads();
  }
}

// ---------------- bbs whitening loss --------------------------------------------
__global__ __launch_bounds__(256) void k_bbsloss(float* __restrict__ ws) {
  const int b = blockIdx.x * blockDim.x + threadIdx.x;
  if (b >= B_DIM) return;
  float inv[CB];
  #pragma unroll
  for (int p = 0; p < CB; ++p) inv[p] = 1.0f / fmaxf(sqrtf(ws[WS_SSP + p]), 1e-6f);
  float s = 0.0f;
  for (int p = 0; p < 45; ++p) {
    int c, d; pair10(p, c, d);
    s += fabsf(ws[WS_COVB + b * 45 + p] * inv[c] * inv[d]);
  }
  s = s / (float)(THW - 1) / 45.0f;
  atomicAdd(&ws[WS_LOSS], s / (float)B_DIM);
}

// ---------------- in-place column scale of bbs_out -------------------------------
__global__ __launch_bounds__(256) void k_scale(float* __restrict__ out,
                                               const float* __restrict__ ws) {
  __shared__ float s_inv[CB];
  if (threadIdx.x < CB)
    s_inv[threadIdx.x] = 1.0f / fmaxf(sqrtf(ws[WS_SSP + threadIdx.x]), 1e-6f);
  __syncthreads();
  const size_t i = (size_t)blockIdx.x * 256 + threadIdx.x;
  if (i < (size_t)B_DIM * CB * THW) {
    const int p = (int)((i / THW) % CB);
    out[i] *= s_inv[p];
  }
}

// ---------------- label-similarity loss: one wave per feature row ----------------
__global__ __launch_bounds__(256) void k_sim(const float* __restrict__ xA,
                                             const float* __restrict__ xB,
                                             const int* __restrict__ labA,
                                             const int* __restrict__ labB,
                                             float* __restrict__ ws) {
  __shared__ float bacc;
  if (threadIdx.x == 0) bacc = 0.0f;
  __syncthreads();
  const int wv = threadIdx.x >> 5, lane = threadIdx.x & 31;
  const int row = blockIdx.x * 8 + wv;
  if (row < 2 * NROW) {
    const float* f; int lab;
    if (row < NROW) { f = xA + (size_t)row * FEA;           lab = labA[row]; }
    else            { f = xB + (size_t)(row - NROW) * FEA;  lab = labB[row - NROW]; }
    const float* cn = ws + WS_CN + lab * FEA;
    float ff = 0.0f, fc = 0.0f;
    #pragma unroll
    for (int q = 0; q < 4; ++q) {
      const float fv = f[lane + 32 * q];
      ff += fv * fv;
      fc += fv * cn[lane + 32 * q];
    }
    #pragma unroll
    for (int mk = 16; mk >= 1; mk >>= 1) {
      ff += __shfl_xor(ff, mk, 32);
      fc += __shfl_xor(fc, mk, 32);
    }
    if (lane == 0 && lab >= 1) {
      const float s = fc / fmaxf(sqrtf(ff), 1e-12f);
      atomicAdd(&bacc, 1.0f - s);
    }
  }
  __syncthreads();
  if (threadIdx.x == 0) atomicAdd(&ws[WS_LOSS], bacc);
}

// ---------------- write scalar loss ----------------------------------------------
__global__ void k_fin(float* __restrict__ out, const float* __restrict__ ws, int out_size) {
  out[out_size - 1] = ws[WS_LOSS];
}

extern "C" void kernel_launch(void* const* d_in, const int* in_sizes, int n_in,
                              void* d_out, int out_size, void* d_ws, size_t ws_size,
                              hipStream_t stream) {
  const float* x       = (const float*)d_in[0];
  const float* xA      = (const float*)d_in[1];
  const float* xB      = (const float*)d_in[2];
  const float* conv_w  = (const float*)d_in[3];
  const float* gate    = (const float*)d_in[4];
  const float* centerf = (const float*)d_in[5];
  const int*   labA    = (const int*)d_in[6];
  const int*   labB    = (const int*)d_in[7];
  float* out = (float*)d_out;
  float* ws  = (float*)d_ws;
  (void)in_sizes; (void)n_in; (void)ws_size;

  k_init  <<<1, 256, 0, stream>>>(conv_w, gate, centerf, ws);
  k_pass1 <<<B_DIM, 256, 0, stream>>>(x, ws);
  k_dgloss<<<B_DIM / 256, 256, 0, stream>>>(ws);
  k_pass2 <<<B_DIM, 256, 0, stream>>>(x, ws, out);
  k_bbsloss<<<B_DIM / 256, 256, 0, stream>>>(ws);
  const int total = B_DIM * CB * THW;
  k_scale <<<(total + 255) / 256, 256, 0, stream>>>(out, ws);
  k_sim   <<<(2 * NROW) / 8, 256, 0, stream>>>(xA, xB, labA, labB, ws);
  k_fin   <<<1, 1, 0, stream>>>(out, ws, out_size);
}
